// RegionNonLocal3D_7481833029822
// MI455X (gfx1250) — matrix operand order repaired and checked by reading, not compiled, not run on hardware
//
#include <hip/hip_runtime.h>
#include <hip/hip_bf16.h>

typedef _Float16 h16;
typedef __attribute__((ext_vector_type(2)))  _Float16 h2;
typedef __attribute__((ext_vector_type(8)))  _Float16 v8h;
typedef __attribute__((ext_vector_type(16))) _Float16 v16h;
typedef __attribute__((ext_vector_type(8)))  float    v8f;
typedef __attribute__((ext_vector_type(4)))  unsigned int v4u;
typedef __attribute__((ext_vector_type(8)))  int      v8i;
typedef __attribute__((ext_vector_type(4)))  int      v4i;

#define HW   2304   // 48*48
#define NC   128    // C
#define NH   64     // h = C/2
#define NT   5
#define NTC  4
#define NPIX 48

#define LDS_ROW 17  // 16 dwords payload + 1 pad dword (bank-conflict-free halves)

// ---------------------------------------------------------------------------
// Kernel 1: projection GEMMs.  out[h,pix] = W[h,:] . x[:,pix]  per (b, t).
// The 128x16 x-tile is DMA'd into LDS once per block by the Tensor Data
// Mover (TDM), with D# padding of 1 dword per 16-dword row (LDS row stride
// 17) so the two lane-halves of each wave hit disjoint bank sets.
// t==0 -> theta (w_theta); t>=1 -> phi (w_phi) and g (w_g).
// One block = 4 waves; wave w owns M rows [16w, 16w+16) of the 64 channels.
// K = 128 -> 4 WMMA 16x16x32 f16 steps.  Stores f16 [pix][64].
// ---------------------------------------------------------------------------
__global__ __launch_bounds__(128) void rnl_proj_kernel(
    const float* __restrict__ x,
    const float* __restrict__ w_theta,
    const float* __restrict__ w_phi,
    const float* __restrict__ w_g,
    h16* __restrict__ theta16,
    h16* __restrict__ phi16,
    h16* __restrict__ g16)
{
  __shared__ float xs[NC * LDS_ROW + 16];   // 128 rows x (16+1) dwords

  const int lane  = threadIdx.x & 31;
  const int wv    = threadIdx.x >> 5;
  const int ntile = blockIdx.x;          // 0..143  (2304/16)
  const int t     = blockIdx.y;          // 0..4
  const int b     = blockIdx.z;          // 0..1
  const int n     = lane & 15;
  const int half  = lane >> 4;
  const int pix   = ntile * 16 + n;
  const int m0    = wv * 16;

  // ---- TDM: DMA the 2-D x tile (16 pixels x 128 channels, f32) to LDS ----
  if (wv == 0) {
    const unsigned long long ga =
        (unsigned long long)(const void*)(x + ((size_t)(b * NC) * NT + t) * HW
                                          + (size_t)ntile * 16);
    const unsigned lds_off = (unsigned)(size_t)(&xs[0]);

    v4u g0;
    g0[0] = 1u;                                      // count=1 (user D#)
    g0[1] = lds_off;                                 // lds_addr (bytes)
    g0[2] = (unsigned)(ga & 0xFFFFFFFFull);          // global_addr[31:0]
    g0[3] = (unsigned)((ga >> 32) & 0x01FFFFFFull)   // global_addr[56:32]
          | (2u << 30);                              // type = 2 ("image")

    v8i g1;
    g1[0] = (2 << 16)        // data_size = 4 bytes
          | (1 << 20)        // pad_enable (load only)
          | (3 << 22);       // pad_interval code 3 = every 16 dwords
                             // pad_amount 0 = 1 dword; workgroup_mask = 0
    g1[1] = (16 << 16);      // tensor_dim0 = 16       (bits 79:48)
    g1[2] = (NC << 16);      // tensor_dim1 = 128      (bits 111:80)
    g1[3] = (16 << 16);      // tile_dim0  = 16        (bits 127:112)
    g1[4] = NC;              // tile_dim1  = 128, tile_dim2 = 0
    g1[5] = NT * HW;         // tensor_dim0_stride = 11520 elements (bits 207:160)
    g1[6] = 0;               // stride hi / tensor_dim1_stride (unused, 2-D)
    g1[7] = 0;

    const v4i gz4 = {0, 0, 0, 0};                    // groups 2/3: 2-D tensor
    const v8i gz8 = {0, 0, 0, 0, 0, 0, 0, 0};        // extra group (clang-23 form)
    __builtin_amdgcn_tensor_load_to_lds(g0, g1, gz4, gz4, gz8, 0);
    __builtin_amdgcn_s_wait_tensorcnt(0);            // issuing wave owns TENSORcnt
  }
  __syncthreads();                                   // publish LDS tile

  // ---- B-matrix fragments from LDS (shared by both GEMMs of this frame) ----
  // Contiguous K order (same as A): VGPR j, lane-half hl: K = hl*16 + {2j, 2j+1}
  v16h bm[4];
  #pragma unroll
  for (int ks = 0; ks < 4; ++ks) {
    const int kb = ks * 32 + half * 16;
    #pragma unroll
    for (int e = 0; e < 16; ++e)
      bm[ks][e] = (_Float16)xs[(kb + e) * LDS_ROW + n];
  }

  const int nW = (t == 0) ? 1 : 2;
  for (int wsel = 0; wsel < nW; ++wsel) {
    const float* W = (t == 0) ? w_theta : (wsel == 0 ? w_phi : w_g);

    v8f acc = {0.f, 0.f, 0.f, 0.f, 0.f, 0.f, 0.f, 0.f};
    #pragma unroll
    for (int ks = 0; ks < 4; ++ks) {
      // A layout: lane-half hl: elems 0..15 -> K = hl*16+{0..15}
      const float* wr = W + (size_t)(m0 + n) * NC + ks * 32 + half * 16;
      v16h a;
      #pragma unroll
      for (int e = 0; e < 8; ++e) {
        a[e]     = (_Float16)wr[e];
        a[e + 8] = (_Float16)wr[8 + e];
      }
      acc = __builtin_amdgcn_wmma_f32_16x16x32_f16(
          false, a, false, bm[ks], (short)0, acc, false, false);
    }

    // D layout: VGPR r, lane-half hl: M = hl*8 + r, N = lane&15
    h16* dst;
    if (t == 0)
      dst = theta16 + ((size_t)(b * HW + pix)) * NH;
    else {
      h16* base = (wsel == 0) ? phi16 : g16;
      dst = base + ((size_t)((b * NTC + (t - 1)) * HW + pix)) * NH;
    }
    const int mrow = m0 + half * 8;
    v8h st;
    #pragma unroll
    for (int r = 0; r < 8; ++r) st[r] = (_Float16)acc[r];
    *(v8h*)(dst + mrow) = st;     // 16B-aligned (mrow % 8 == 0)
  }
}

// ---------------------------------------------------------------------------
// Kernel 2: per-pixel neighborhood attention (one wave32 per pixel).
// Lane l owns channels {2l, 2l+1}.  196 scores -> softmax -> y[64] (f16).
// ---------------------------------------------------------------------------
__global__ __launch_bounds__(256) void rnl_attn_kernel(
    const h16* __restrict__ theta16,
    const h16* __restrict__ phi16,
    const h16* __restrict__ g16,
    h16* __restrict__ y16)
{
  const int lane = threadIdx.x & 31;
  const int wid  = blockIdx.x * (blockDim.x >> 5) + (threadIdx.x >> 5); // pixel id
  const int b    = wid / HW;
  const int pix  = wid - b * HW;
  const int iy   = pix / NPIX;
  const int ix   = pix - iy * NPIX;

  const h2 qp = *(const h2*)(theta16 + (size_t)wid * NH + 2 * lane);
  const float q0 = (float)qp[0], q1 = (float)qp[1];

  float sc[7];
  #pragma unroll
  for (int kk = 0; kk < 7; ++kk) sc[kk] = -3.0e38f;

  // ---- pass 1: scores (dot over h via wave reduction) ----
  #pragma unroll
  for (int kk = 0; kk < 7; ++kk) {
    const int kmax = (kk == 6) ? 4 : 32;     // 196 = 6*32 + 4
    for (int l = 0; l < kmax; ++l) {
      const int k  = kk * 32 + l;
      const int t  = k / 49;
      const int r  = k - t * 49;
      const int dy = r / 7;
      const int dx = r - dy * 7;
      int yy = iy + dy - 3; yy = yy < 0 ? 0 : (yy > NPIX - 1 ? NPIX - 1 : yy);
      int xx = ix + dx - 3; xx = xx < 0 ? 0 : (xx > NPIX - 1 ? NPIX - 1 : xx);
      const size_t nb = ((size_t)(b * NTC + t) * HW + yy * NPIX + xx) * NH;
      const h2 pp = *(const h2*)(phi16 + nb + 2 * lane);
      float part = q0 * (float)pp[0] + q1 * (float)pp[1];
      #pragma unroll
      for (int off = 16; off > 0; off >>= 1)
        part += __shfl_xor(part, off, 32);
      if (lane == l) sc[kk] = part * 8.0f;   // * sqrt(h)
    }
  }

  // ---- softmax over 196 (distributed 7 slots/lane) ----
  float m = sc[0];
  #pragma unroll
  for (int kk = 1; kk < 7; ++kk) m = fmaxf(m, sc[kk]);
  #pragma unroll
  for (int off = 16; off > 0; off >>= 1)
    m = fmaxf(m, __shfl_xor(m, off, 32));

  float aw[7];
  float s = 0.f;
  #pragma unroll
  for (int kk = 0; kk < 7; ++kk) { aw[kk] = __expf(sc[kk] - m); s += aw[kk]; }
  #pragma unroll
  for (int off = 16; off > 0; off >>= 1)
    s += __shfl_xor(s, off, 32);
  const float invs = 1.0f / s;

  // ---- pass 2: y = sum_k a_k * g_k ----
  float y0 = 0.f, y1 = 0.f;
  #pragma unroll
  for (int kk = 0; kk < 7; ++kk) {
    const int kmax = (kk == 6) ? 4 : 32;
    const float awn = aw[kk] * invs;
    for (int l = 0; l < kmax; ++l) {
      const int k  = kk * 32 + l;
      const int t  = k / 49;
      const int r  = k - t * 49;
      const int dy = r / 7;
      const int dx = r - dy * 7;
      int yy = iy + dy - 3; yy = yy < 0 ? 0 : (yy > NPIX - 1 ? NPIX - 1 : yy);
      int xx = ix + dx - 3; xx = xx < 0 ? 0 : (xx > NPIX - 1 ? NPIX - 1 : xx);
      const size_t nb = ((size_t)(b * NTC + t) * HW + yy * NPIX + xx) * NH;
      const float av = __shfl(awn, l, 32);
      const h2 gp = *(const h2*)(g16 + nb + 2 * lane);
      y0 += av * (float)gp[0];
      y1 += av * (float)gp[1];
    }
  }

  h2 yo; yo[0] = (_Float16)y0; yo[1] = (_Float16)y1;
  *(h2*)(y16 + (size_t)wid * NH + 2 * lane) = yo;
}

// ---------------------------------------------------------------------------
// Kernel 3: output projection GEMM + residual.
// out[pix, c] = x0[c, pix] + sum_h y[pix, h] * w_out[c, h]
// Wave computes one 16-pix x 16-chan tile; K = 64 -> 2 WMMA steps.
// ---------------------------------------------------------------------------
__global__ __launch_bounds__(128) void rnl_out_kernel(
    const h16*  __restrict__ y16,
    const float* __restrict__ w_out,
    const float* __restrict__ x,
    float* __restrict__ out)
{
  const int lane  = threadIdx.x & 31;
  const int wv    = threadIdx.x >> 5;
  const int ptile = blockIdx.x;            // 0..287 over B*HW/16
  const int ctile = blockIdx.y * 4 + wv;   // 0..7
  const int n     = lane & 15;
  const int half  = lane >> 4;
  const int c     = ctile * 16 + n;

  v8f acc = {0.f, 0.f, 0.f, 0.f, 0.f, 0.f, 0.f, 0.f};
  #pragma unroll
  for (int ks = 0; ks < 2; ++ks) {
    // A = y rows (pixels), already f16, contiguous 16B chunks
    const h16* ya = y16 + (size_t)(ptile * 16 + n) * NH + ks * 32 + half * 16;
    const v8h lo = *(const v8h*)(ya);
    const v8h hi = *(const v8h*)(ya + 8);
    v16h a;
    #pragma unroll
    for (int e = 0; e < 8; ++e) { a[e] = lo[e]; a[e + 8] = hi[e]; }

    // B[k][n] = w_out[c=n][h=k] ; lane-half hl: K = hl*16 + e
    const float* wr = w_out + (size_t)c * NH + ks * 32 + half * 16;
    v16h bm;
    #pragma unroll
    for (int e = 0; e < 16; ++e) bm[e] = (_Float16)wr[e];

    acc = __builtin_amdgcn_wmma_f32_16x16x32_f16(
        false, a, false, bm, (short)0, acc, false, false);
  }

  // D: M = pixel offset = half*8 + r, N = channel
  const int prow    = ptile * 16 + half * 8;       // global pixel row base
  const int b       = prow / HW;
  const int pixloc  = prow - b * HW;
  const float* xi   = x   + ((size_t)(b * NC + c) * NT) * HW + pixloc; // t=0
  float*       op   = out + ((size_t)(b * NC + c)) * HW + pixloc;
  #pragma unroll
  for (int r = 0; r < 8; ++r) op[r] = acc[r] + xi[r];
}

// ---------------------------------------------------------------------------
extern "C" void kernel_launch(void* const* d_in, const int* in_sizes, int n_in,
                              void* d_out, int out_size, void* d_ws, size_t ws_size,
                              hipStream_t stream) {
  const float* x       = (const float*)d_in[0];
  const float* w_theta = (const float*)d_in[1];
  const float* w_phi   = (const float*)d_in[2];
  const float* w_g     = (const float*)d_in[3];
  const float* w_out   = (const float*)d_in[4];
  float* out = (float*)d_out;

  // Workspace layout (f16):
  //   theta16 : [B*HW][64]        = 589,824 B
  //   phi16   : [B*4*HW][64]      = 2,359,296 B
  //   g16     : [B*4*HW][64]      = 2,359,296 B
  //   y16     : [B*HW][64]        = 589,824 B        (total ~5.9 MB)
  char* ws = (char*)d_ws;
  h16* theta16 = (h16*)(ws);
  h16* phi16   = (h16*)(ws + 589824);
  h16* g16     = (h16*)(ws + 589824 + 2359296);
  h16* y16     = (h16*)(ws + 589824 + 2 * 2359296);

  dim3 g1(HW / 16, NT, 2);                 // 144 x 5 x 2, 4 waves/block
  rnl_proj_kernel<<<g1, 128, 0, stream>>>(x, w_theta, w_phi, w_g,
                                          theta16, phi16, g16);

  rnl_attn_kernel<<<(2 * HW) / 8, 256, 0, stream>>>(theta16, phi16, g16, y16);

  dim3 g3((2 * HW) / 16, 2);               // 288 pixel tiles x 2 (4 waves -> 8 ctiles)
  rnl_out_kernel<<<g3, 128, 0, stream>>>(y16, w_out, x, out);
}
